// DCNRefine3D_Enhanced_79860621902197
// MI455X (gfx1250) — matrix operand order repaired and checked by reading, not compiled, not run on hardware
//
#include <hip/hip_runtime.h>
#include <hip/hip_bf16.h>

// ---------------------------------------------------------------------------
// Problem constants (from reference)
// ---------------------------------------------------------------------------
#define NB   2
#define CCH  64
#define DDim 8
#define HDim 96
#define WDim 96
#define DHW  (DDim * HDim * WDim)   // 73728
#define NDHW (NB * DHW)             // 147456
#define GRP  2
#define PPTS 27
#define CGC  32

typedef __bf16 bf16;
typedef __attribute__((ext_vector_type(16))) __bf16 v16bf;
typedef __attribute__((ext_vector_type(8)))  float  v8f;

// ---------------------------------------------------------------------------
// WMMA helpers (CDNA5 v_wmma_f32_16x16x32_bf16; layouts per ISA 7.12.2)
// ---------------------------------------------------------------------------
__device__ __forceinline__ v8f wmma_bf16(v16bf a, v16bf b, v8f c) {
    // 8 args: (neg_a, A, neg_b, B, c_mod, C, reuse_a, reuse_b)
    return __builtin_amdgcn_wmma_f32_16x16x32_bf16(false, a, false, b,
                                                   (short)0, c, false, false);
}

// A-matrix 16x32 bf16 fragment from f32 LDS tile (row-major, leading dim ld).
// Lane L: m = L&15, hi = L>>4; element i -> K = kb*32 + hi*8 + (i<8 ? i : i+8)
__device__ __forceinline__ v16bf a_frag_f32(const float* s, int ld, int rowBase, int kb) {
    int lane = threadIdx.x & 31;
    int m = lane & 15, hi = lane >> 4;
    const float* row = s + (rowBase + m) * ld + kb * 32 + hi * 8;
    v16bf a;
#pragma unroll
    for (int i = 0; i < 16; ++i) a[i] = (bf16)row[(i < 8) ? i : (i + 8)];
    return a;
}

__device__ __forceinline__ v16bf a_frag_bf16(const bf16* s, int ld, int rowBase, int kb) {
    int lane = threadIdx.x & 31;
    int m = lane & 15, hi = lane >> 4;
    const bf16* row = s + (rowBase + m) * ld + kb * 32 + hi * 8;
    v16bf a;
#pragma unroll
    for (int i = 0; i < 16; ++i) a[i] = row[(i < 8) ? i : (i + 8)];
    return a;
}

// B-matrix 32x16 bf16 fragment from bf16 LDS weight panel.
// transp=false: B[k][n] = w[k*ld + n]; transp=true: B[k][n] = w[n*ld + k].
// Lane L: n = nb*16 + (L&15); element i -> k = kb*32 + (L>>4)*8 + (i<8 ? i : i+8)
__device__ __forceinline__ v16bf b_frag(const bf16* w, int ld, int nb, int kb, bool transp) {
    int lane = threadIdx.x & 31;
    int n = nb * 16 + (lane & 15);
    int k0 = kb * 32 + (lane >> 4) * 8;
    v16bf b;
#pragma unroll
    for (int i = 0; i < 16; ++i) {
        int k = k0 + ((i < 8) ? i : (i + 8));
        b[i] = transp ? w[n * ld + k] : w[k * ld + n];
    }
    return b;
}

// D (16x16 f32) stage to LDS: lane L holds col n=L&15, rows (L>>4)*8 + r
__device__ __forceinline__ void d_stage(float* s, int ld, int rowBase, int colBase, v8f d) {
    int lane = threadIdx.x & 31;
    int n = colBase + (lane & 15);
    int m0 = (lane >> 4) * 8;
#pragma unroll
    for (int r = 0; r < 8; ++r) s[(rowBase + m0 + r) * ld + n] = d[r];
}

__device__ __forceinline__ float gelu_tanh(float x) {
    float x3 = x * x * x;
    return 0.5f * x * (1.f + tanhf(0.7978845608028654f * (x + 0.044715f * x3)));
}

// ---------------------------------------------------------------------------
// K1: pre = x · w_pre^T ; x_proj = pre · w_in + b_in  (channels-last outputs)
// ---------------------------------------------------------------------------
__global__ __launch_bounds__(256) void k_pre_proj(
    const float* __restrict__ x, const float* __restrict__ w_pre,
    const float* __restrict__ w_in, const float* __restrict__ b_in,
    float* __restrict__ pre_cl, float* __restrict__ xproj_cl)
{
    __shared__ float S[128 * 65];
    __shared__ bf16 sW0[64 * 64];
    __shared__ bf16 sW1[64 * 64];
    const int tid = threadIdx.x;
    const int tile = blockIdx.x * 128;

    for (int i = tid; i < 64 * 64; i += 256) {
        sW0[i] = (bf16)w_pre[i];
        sW1[i] = (bf16)w_in[i];
    }
    // Stage x tile (NCDHW -> [vox][ch]), coalesced along contiguous W runs.
    for (int e = tid; e < 128 * 64; e += 256) {
        int c = e >> 7, vs = e & 127;
        int v = tile + vs;
        int n = v / DHW, r = v % DHW;
        S[vs * 65 + c] = x[((size_t)(n * CCH + c)) * DHW + r];
    }
    __syncthreads();

    const int wid = tid >> 5;
    const int rowBase = wid * 16;

    v16bf a0 = a_frag_f32(S, 65, rowBase, 0);
    v16bf a1 = a_frag_f32(S, 65, rowBase, 1);
    v8f acc[4];
#pragma unroll
    for (int nt = 0; nt < 4; ++nt) {
        v8f c = {};
        c = wmma_bf16(a0, b_frag(sW0, 64, nt, 0, true), c);   // B = w_pre^T
        c = wmma_bf16(a1, b_frag(sW0, 64, nt, 1, true), c);
        acc[nt] = c;
    }
    __syncthreads();
#pragma unroll
    for (int nt = 0; nt < 4; ++nt) d_stage(S, 65, rowBase, nt * 16, acc[nt]);
    __syncthreads();

    // Write pre (channels-last, contiguous) and rebuild A-frags of pre
    for (int e = tid; e < 128 * 64; e += 256) {
        int vs = e >> 6, c = e & 63;
        pre_cl[(size_t)tile * 64 + e] = S[vs * 65 + c];
    }
    a0 = a_frag_f32(S, 65, rowBase, 0);
    a1 = a_frag_f32(S, 65, rowBase, 1);
#pragma unroll
    for (int nt = 0; nt < 4; ++nt) {
        v8f c = {};
        c = wmma_bf16(a0, b_frag(sW1, 64, nt, 0, false), c);  // B = w_in
        c = wmma_bf16(a1, b_frag(sW1, 64, nt, 1, false), c);
        acc[nt] = c;
    }
    __syncthreads();
#pragma unroll
    for (int nt = 0; nt < 4; ++nt) d_stage(S, 65, rowBase, nt * 16, acc[nt]);
    __syncthreads();
    for (int e = tid; e < 128 * 64; e += 256) {
        int vs = e >> 6, c = e & 63;
        xproj_cl[(size_t)tile * 64 + e] = S[vs * 65 + c] + b_in[c];
    }
}

// ---------------------------------------------------------------------------
// K2: depthwise 3x3x3 conv on channels-last pre (zero pad = 1)
// ---------------------------------------------------------------------------
__global__ __launch_bounds__(256) void k_dw(
    const float* __restrict__ pre_cl, const float* __restrict__ w_dw,
    float* __restrict__ dw_cl)
{
    __shared__ float sw[64 * 27];
    const int tid = threadIdx.x;
    for (int i = tid; i < 64 * 27; i += 256) sw[i] = w_dw[i]; // [c][1][3][3][3]
    __syncthreads();

    const int c = tid & 63;
    const int vs = tid >> 6;
    const int v = blockIdx.x * 4 + vs;
    const int n = v / DHW, r = v % DHW;
    const int d = r / (HDim * WDim);
    const int r2 = r % (HDim * WDim);
    const int h = r2 / WDim, w = r2 % WDim;

    float acc = 0.f;
#pragma unroll
    for (int kz = 0; kz < 3; ++kz) {
        int dz = d + kz - 1;
        if (dz < 0 || dz >= DDim) continue;
#pragma unroll
        for (int ky = 0; ky < 3; ++ky) {
            int hy = h + ky - 1;
            if (hy < 0 || hy >= HDim) continue;
#pragma unroll
            for (int kx = 0; kx < 3; ++kx) {
                int wx = w + kx - 1;
                if (wx < 0 || wx >= WDim) continue;
                size_t vn = (size_t)((n * DDim + dz) * HDim + hy) * WDim + wx;
                acc += pre_cl[vn * 64 + c] * sw[c * 27 + kz * 9 + ky * 3 + kx];
            }
        }
    }
    dw_cl[(size_t)v * 64 + c] = acc;
}

// ---------------------------------------------------------------------------
// K2b/K2c: deterministic instance-norm statistics (sum, sumsq per (n,c))
// ---------------------------------------------------------------------------
__global__ __launch_bounds__(256) void k_stats1(
    const float* __restrict__ dw_cl, float* __restrict__ part)
{
    const int n = blockIdx.y, chunk = blockIdx.x;      // 64 chunks per n
    const int tid = threadIdx.x;
    const int c = tid & 63, q = tid >> 6;              // q in 0..3
    const int CH = DHW / 64;                           // 1152 voxels/chunk
    const int v0 = n * DHW + chunk * CH + q * (CH / 4);

    float s = 0.f, ss = 0.f;
    for (int i = 0; i < CH / 4; ++i) {
        float xv = dw_cl[(size_t)(v0 + i) * 64 + c];
        s += xv;
        ss += xv * xv;
    }
    __shared__ float ps[256], pss[256];
    ps[tid] = s; pss[tid] = ss;
    __syncthreads();
    if (q == 0) {
        float S = ps[c] + ps[64 + c] + ps[128 + c] + ps[192 + c];
        float SS = pss[c] + pss[64 + c] + pss[128 + c] + pss[192 + c];
        part[((size_t)(n * 64 + chunk) * 64 + c) * 2 + 0] = S;
        part[((size_t)(n * 64 + chunk) * 64 + c) * 2 + 1] = SS;
    }
}

__global__ __launch_bounds__(128) void k_stats2(
    const float* __restrict__ part, float* __restrict__ stats)
{
    const int tid = threadIdx.x;
    if (tid >= NB * 64) return;
    const int n = tid >> 6, c = tid & 63;
    float S = 0.f, SS = 0.f;
    for (int ch = 0; ch < 64; ++ch) {
        S += part[((size_t)(n * 64 + ch) * 64 + c) * 2 + 0];
        SS += part[((size_t)(n * 64 + ch) * 64 + c) * 2 + 1];
    }
    float inv = 1.f / (float)DHW;
    float mean = S * inv;
    float var = SS * inv - mean * mean;
    stats[(n * 64 + c) * 2 + 0] = mean;
    stats[(n * 64 + c) * 2 + 1] = rsqrtf(var + 1e-5f);
}

// ---------------------------------------------------------------------------
// K3: fused IN+GELU, offset/mask WMMA GEMMs, softmax, trilinear gather core.
// Block = 64 threads (2 waves), tile = 32 voxels (16 per wave).
// ---------------------------------------------------------------------------
__global__ __launch_bounds__(64) void k_dcn(
    const float* __restrict__ dw_cl, const float* __restrict__ stats,
    const float* __restrict__ w_off, const float* __restrict__ b_off,
    const float* __restrict__ w_mask, const float* __restrict__ b_mask,
    const float* __restrict__ xproj_cl, float* __restrict__ core_cl)
{
    __shared__ bf16 s_feat[32 * 64];           // 4 KB
    __shared__ bf16 s_w[64 * 240];             // 30 KB: [c][0..175]=w_off pad, [176..239]=w_mask pad
    __shared__ float s_off[2][16 * 164];       // 21 KB (cols 0..161 valid)
    __shared__ float s_mask[2][16 * 56];       // 7 KB  ([m][g*28+p])
    const int tid = threadIdx.x;
    const int tile = blockIdx.x * 32;
    const int n = tile / DHW;

    for (int i = tid; i < 64 * 240; i += 64) {
        int rr = i / 240, cc = i % 240;
        float v;
        if (cc < 176) v = (cc < 162) ? w_off[rr * 162 + cc] : 0.f;
        else { int c2 = cc - 176; v = (c2 < 54) ? w_mask[rr * 54 + c2] : 0.f; }
        s_w[i] = (bf16)v;
    }
    for (int i = tid; i < 32 * 64; i += 64) {
        int vs = i >> 6, c = i & 63;
        float xv = dw_cl[(size_t)(tile + vs) * 64 + c];
        float mean = stats[(n * 64 + c) * 2 + 0];
        float rstd = stats[(n * 64 + c) * 2 + 1];
        s_feat[vs * 64 + c] = (bf16)gelu_tanh((xv - mean) * rstd);
    }
    __syncthreads();

    const int wid = tid >> 5;
    const int lane = tid & 31;
    const int rowBase = wid * 16;

    v16bf a0 = a_frag_bf16(s_feat, 64, rowBase, 0);
    v16bf a1 = a_frag_bf16(s_feat, 64, rowBase, 1);

    // offset GEMM: 162 outputs -> 11 N-tiles of 16
    for (int nt = 0; nt < 11; ++nt) {
        v8f c = {};
        c = wmma_bf16(a0, b_frag(s_w, 240, nt, 0, false), c);
        c = wmma_bf16(a1, b_frag(s_w, 240, nt, 1, false), c);
        int col = nt * 16 + (lane & 15);
        int m0 = (lane >> 4) * 8;
        if (col < 162) {
            float bo = b_off[col];
#pragma unroll
            for (int r = 0; r < 8; ++r) s_off[wid][(m0 + r) * 164 + col] = c[r] + bo;
        }
    }
    // mask GEMM: 54 outputs -> 4 N-tiles starting at panel col 176 (= nb 11)
    for (int nt = 0; nt < 4; ++nt) {
        v8f c = {};
        c = wmma_bf16(a0, b_frag(s_w, 240, 11 + nt, 0, false), c);
        c = wmma_bf16(a1, b_frag(s_w, 240, 11 + nt, 1, false), c);
        int col = nt * 16 + (lane & 15);
        int m0 = (lane >> 4) * 8;
        if (col < 54) {
            float bm = b_mask[col];
            int g = col / 27, p = col % 27;
#pragma unroll
            for (int r = 0; r < 8; ++r) s_mask[wid][(m0 + r) * 56 + g * 28 + p] = c[r] + bm;
        }
    }
    __syncthreads();

    // softmax over P=27; lane handles (m = lane&15, g = lane>>4)
    {
        int m = lane & 15, g = lane >> 4;
        float* ml = &s_mask[wid][m * 56 + g * 28];
        float mx = -1e30f;
#pragma unroll
        for (int p = 0; p < PPTS; ++p) mx = fmaxf(mx, ml[p]);
        float sum = 0.f;
#pragma unroll
        for (int p = 0; p < PPTS; ++p) { float e = __expf(ml[p] - mx); ml[p] = e; sum += e; }
        float inv = 1.f / sum;
#pragma unroll
        for (int p = 0; p < PPTS; ++p) ml[p] *= inv;
    }
    __syncthreads();

    // gather core: lane = channel within group; 128B coalesced corner loads
    for (int g = 0; g < GRP; ++g) {
        for (int m = 0; m < 16; ++m) {
            int v = tile + rowBase + m;
            int r = v - n * DHW;
            int d = r / (HDim * WDim);
            int r2 = r % (HDim * WDim);
            int h = r2 / WDim, w = r2 % WDim;
            const float* offrow = &s_off[wid][m * 164 + g * 81];
            const float* mrow = &s_mask[wid][m * 56 + g * 28];
            float a = 0.f;
            for (int p = 0; p < PPTS; ++p) {
                int kz = p / 9, ky = (p % 9) / 3, kx = p % 3;
                float ox = 0.5f * offrow[p * 3 + 0];   // AXIS_SCALE dx
                float oy = offrow[p * 3 + 1];
                float oz = offrow[p * 3 + 2];
                float zs = (float)(d + kz) + oz;       // padded-frame coords
                float ys = (float)(h + ky) + oy;
                float xs = (float)(w + kx) + ox;
                float z0f = floorf(zs), y0f = floorf(ys), x0f = floorf(xs);
                float fz = zs - z0f, fy = ys - y0f, fx = xs - x0f;
                int z0 = (int)z0f - 1, y0 = (int)y0f - 1, x0 = (int)x0f - 1; // unpadded
                float wz[2] = {1.f - fz, fz};
                float wy[2] = {1.f - fy, fy};
                float wx[2] = {1.f - fx, fx};
                float samp = 0.f;
#pragma unroll
                for (int cz = 0; cz < 2; ++cz) {
                    int zc = z0 + cz;
                    if ((unsigned)zc >= (unsigned)DDim) continue;
#pragma unroll
                    for (int cy = 0; cy < 2; ++cy) {
                        int yc = y0 + cy;
                        if ((unsigned)yc >= (unsigned)HDim) continue;
#pragma unroll
                        for (int cx = 0; cx < 2; ++cx) {
                            int xc = x0 + cx;
                            if ((unsigned)xc >= (unsigned)WDim) continue;
                            size_t vi = (size_t)((n * DDim + zc) * HDim + yc) * WDim + xc;
                            samp += (wz[cz] * wy[cy] * wx[cx]) *
                                    xproj_cl[vi * 64 + g * CGC + lane];
                        }
                    }
                }
                a += mrow[p] * samp;
            }
            core_cl[(size_t)v * 64 + g * CGC + lane] = a;
        }
    }
}

// ---------------------------------------------------------------------------
// K4: out = core · w_out + b_out ; y = out · w_post^T ; result = x + sig(gate)*y
// ---------------------------------------------------------------------------
__global__ __launch_bounds__(256) void k_out_post(
    const float* __restrict__ core_cl, const float* __restrict__ w_out,
    const float* __restrict__ b_out, const float* __restrict__ w_post,
    const float* __restrict__ x, const float* __restrict__ gate,
    float* __restrict__ out)
{
    __shared__ float S[128 * 65];
    __shared__ bf16 sW0[64 * 64];
    __shared__ bf16 sW1[64 * 64];
    const int tid = threadIdx.x;
    const int tile = blockIdx.x * 128;

    for (int i = tid; i < 64 * 64; i += 256) {
        sW0[i] = (bf16)w_out[i];
        sW1[i] = (bf16)w_post[i];
    }
    for (int e = tid; e < 128 * 64; e += 256) {
        int vs = e >> 6, c = e & 63;
        S[vs * 65 + c] = core_cl[(size_t)tile * 64 + e];
    }
    __syncthreads();

    const int wid = tid >> 5;
    const int lane = tid & 31;
    const int rowBase = wid * 16;

    v16bf a0 = a_frag_f32(S, 65, rowBase, 0);
    v16bf a1 = a_frag_f32(S, 65, rowBase, 1);
    v8f acc[4];
#pragma unroll
    for (int nt = 0; nt < 4; ++nt) {
        v8f c = {};
        c = wmma_bf16(a0, b_frag(sW0, 64, nt, 0, false), c);  // B = w_out
        c = wmma_bf16(a1, b_frag(sW0, 64, nt, 1, false), c);
        acc[nt] = c;
    }
    __syncthreads();
#pragma unroll
    for (int nt = 0; nt < 4; ++nt) {
        int col = nt * 16 + (lane & 15);
        int m0 = (lane >> 4) * 8;
        float bo = b_out[col];
#pragma unroll
        for (int r = 0; r < 8; ++r) S[(rowBase + m0 + r) * 65 + col] = acc[nt][r] + bo;
    }
    __syncthreads();

    a0 = a_frag_f32(S, 65, rowBase, 0);
    a1 = a_frag_f32(S, 65, rowBase, 1);
#pragma unroll
    for (int nt = 0; nt < 4; ++nt) {
        v8f c = {};
        c = wmma_bf16(a0, b_frag(sW1, 64, nt, 0, true), c);   // B = w_post^T
        c = wmma_bf16(a1, b_frag(sW1, 64, nt, 1, true), c);
        acc[nt] = c;
    }
    __syncthreads();
#pragma unroll
    for (int nt = 0; nt < 4; ++nt) d_stage(S, 65, rowBase, nt * 16, acc[nt]);
    __syncthreads();

    float sg = 1.f / (1.f + __expf(-gate[0]));
    for (int e = tid; e < 128 * 64; e += 256) {
        int c = e >> 7, vs = e & 127;
        int v = tile + vs;
        int nn = v / DHW, r = v % DHW;
        size_t gi = ((size_t)(nn * CCH + c)) * DHW + r;
        out[gi] = x[gi] + sg * S[vs * 65 + c];
    }
}

// ---------------------------------------------------------------------------
// Host entry
// ---------------------------------------------------------------------------
extern "C" void kernel_launch(void* const* d_in, const int* in_sizes, int n_in,
                              void* d_out, int out_size, void* d_ws, size_t ws_size,
                              hipStream_t stream)
{
    const float* x      = (const float*)d_in[0];
    const float* w_pre  = (const float*)d_in[1];
    const float* w_in   = (const float*)d_in[2];
    const float* b_in   = (const float*)d_in[3];
    const float* w_dw   = (const float*)d_in[4];
    const float* w_off  = (const float*)d_in[5];
    const float* b_off  = (const float*)d_in[6];
    const float* w_mask = (const float*)d_in[7];
    const float* b_mask = (const float*)d_in[8];
    const float* w_out  = (const float*)d_in[9];
    const float* b_out  = (const float*)d_in[10];
    const float* w_post = (const float*)d_in[11];
    const float* gate   = (const float*)d_in[12];

    float* ws = (float*)d_ws;
    const size_t T = (size_t)NDHW * 64;
    float* pre_cl   = ws;              // reused as core_cl after k_dw
    float* xproj_cl = ws + T;
    float* dw_cl    = ws + 2 * T;
    float* part     = ws + 3 * T;      // 2*64*64*2 floats
    float* stats    = part + 2 * 64 * 64 * 2;
    float* out      = (float*)d_out;

    k_pre_proj<<<NDHW / 128, 256, 0, stream>>>(x, w_pre, w_in, b_in, pre_cl, xproj_cl);
    k_dw<<<NDHW / 4, 256, 0, stream>>>(pre_cl, w_dw, dw_cl);
    dim3 g2(64, NB);
    k_stats1<<<g2, 256, 0, stream>>>(dw_cl, part);
    k_stats2<<<1, 128, 0, stream>>>(part, stats);
    float* core_cl = pre_cl;           // pre no longer needed
    k_dcn<<<NDHW / 32, 64, 0, stream>>>(dw_cl, stats, w_off, b_off, w_mask, b_mask,
                                        xproj_cl, core_cl);
    k_out_post<<<NDHW / 128, 256, 0, stream>>>(core_cl, w_out, b_out, w_post, x, gate, out);
}
